// PairwiseGeometricNetX_48335561949418
// MI455X (gfx1250) — matrix operand order repaired and checked by reading, not compiled, not run on hardware
//
#include <hip/hip_runtime.h>
#include <hip/hip_bf16.h>

// ---------------------------------------------------------------------------
// PairwiseGeometricNet for MI455X (gfx1250, wave32, WMMA 16x16x32 f16->f32)
//
// Input order assumption: d_in[0..5] = prot_pos, prot_feat, lig_pos, lig_feat,
// prot_mask(u8), lig_mask(u8); then params pytree leaves in sorted-key order.
//
// - B operands staged in LDS in fragment-major order: one WMMA B fragment is a
//   contiguous 32B per-lane load (2x ds_load_b128).
// - The per-layer message tensor t is produced by k_gemm directly in f16
//   fragment-major chunk layout, so k_mp_attn stages it with
//   GLOBAL_LOAD_ASYNC_TO_LDS_B128 (ASYNCcnt) -- no VGPR round trip.
// ---------------------------------------------------------------------------

#define DEV __device__ __forceinline__

typedef __attribute__((ext_vector_type(16))) _Float16 v16h_t;
typedef __attribute__((ext_vector_type(8)))  float    v8f_t;

union AFrag { v16h_t v; _Float16 h[16]; };
union CFrag { v8f_t  v; float     f[8];  };

constexpr int HID   = 160;
constexpr int NFT   = 20;
constexpr int AH    = 40;   // ATTN_H
constexpr int PRD   = 80;   // PAIR
constexpr int MBINS = 20;
constexpr int CBINS = 32;
constexpr int Bn    = 2;
constexpr int Pn    = 1536;
constexpr int Ln    = 128;
constexpr int CH_HALFS = 32 * HID;  // 5120 halfs per 32-row swizzled chunk

DEV float gelu_f(float x) {
    const float c = 0.7978845608028654f; // sqrt(2/pi)
    float t = c * (x + 0.044715f * x * x * x);
    return 0.5f * x * (1.0f + tanhf(t));
}

DEV v8f_t wmma16(const AFrag& a, const AFrag& b, v8f_t c) {
    return __builtin_amdgcn_wmma_f32_16x16x32_f16(
        false, a.v, false, b.v, (short)0, c, false, false);
}

// A fragment: 16x32 f16 tile, row-major in LDS with 'stride' halfs per row.
DEV AFrag load_a16(const _Float16* base, int stride) {
    int lane = threadIdx.x & 31;
    int r = lane & 15, kb = (lane >> 4) * 8;
    const _Float16* p = base + r * stride + kb;
    AFrag f;
    #pragma unroll
    for (int q = 0; q < 8; ++q) { f.h[q] = p[q]; f.h[8 + q] = p[16 + q]; }
    return f;
}

// Fragment-major B: each lane's 16 halfs contiguous (32B) -> 2x ds_load_b128.
DEV AFrag load_bfrag(const _Float16* frag_base) {
    int lane = threadIdx.x & 31;
    AFrag f;
    f.v = *(const v16h_t*)(frag_base + lane * 16);
    return f;
}

// index of element (k, n) inside a fragment-major B buffer with NT n-tiles.
// Fragment (kc, nt) occupies halfs [(kc*NT+nt)*512, +512).
DEV int bswz(int k, int n, int NT) {
    int kc = k >> 5, kl = k & 31, nt = n >> 4, nl = n & 15;
    int lane = (((kl >> 3) & 1) << 4) | nl;
    int q = (kl & 7) | ((kl >> 4) << 3);
    return ((kc * NT + nt) * 32 + lane) * 16 + q;
}

// Async global->LDS 16-byte copy (per-lane addresses), tracked by ASYNCcnt.
DEV void async_copy_b128(uint32_t lds_off, const void* gaddr) {
    asm volatile("global_load_async_to_lds_b128 %0, %1, off"
                 :: "v"(lds_off), "v"(gaddr) : "memory");
}
DEV void async_wait0() {
    asm volatile("s_wait_asynccnt 0x0" ::: "memory");
}

// ---------------------------------------------------------------------------
// Generic GEMM: out[M,N] = act(A[M,K] @ W[K,N] + bias[N]); act 1 -> gelu
// swz=0: f32 row-major out. swz=1 (requires N==160): f16 fragment-major
// 32-row chunks, ready for async staging + load_bfrag in k_mp_attn.
// ---------------------------------------------------------------------------
__global__ __launch_bounds__(256)
void k_gemm(const float* __restrict__ A, const float* __restrict__ W,
            const float* __restrict__ bias, void* __restrict__ outp,
            int M, int K, int N, int act, int swz) {
    __shared__ alignas(32) _Float16 a_s[16 * 160];
    __shared__ alignas(32) _Float16 w_s[160 * 160];
    __shared__ float b_s[160];

    int Kp  = (K + 31) & ~31;
    int Np  = (N + 15) & ~15;
    int NT  = Np >> 4, KC = Kp >> 5;
    int m0  = blockIdx.x * 16;
    int tid = threadIdx.x, lane = tid & 31, wv = tid >> 5;

    for (int e = tid; e < Kp * Np; e += 256) {
        int k = e / Np, n = e % Np;
        w_s[bswz(k, n, NT)] = (_Float16)((k < K && n < N) ? W[k * N + n] : 0.0f);
    }
    for (int e = tid; e < 16 * Kp; e += 256) {
        int r = e / Kp, k = e % Kp;
        int gm = m0 + r;
        a_s[e] = (_Float16)((gm < M && k < K) ? A[gm * K + k] : 0.0f);
    }
    for (int e = tid; e < Np; e += 256) b_s[e] = (e < N) ? bias[e] : 0.0f;
    __syncthreads();

    for (int nt = wv; nt < NT; nt += 8) {
        CFrag c;
        #pragma unroll
        for (int v = 0; v < 8; ++v) c.f[v] = 0.0f;
        for (int kc = 0; kc < KC; ++kc) {
            AFrag a  = load_a16(a_s + kc * 32, Kp);
            AFrag bf = load_bfrag(w_s + (kc * NT + nt) * 512);
            c.v = wmma16(a, bf, c.v);
        }
        int n = nt * 16 + (lane & 15);
        float bb = b_s[n];
        #pragma unroll
        for (int v = 0; v < 8; ++v) {
            int m = m0 + v + 8 * (lane >> 4);
            if (m < M && n < N) {
                float x = c.f[v] + bb;
                if (act) x = gelu_f(x);
                if (!swz) {
                    ((float*)outp)[m * N + n] = x;
                } else {
                    ((_Float16*)outp)[(m >> 5) * CH_HALFS + bswz(m & 31, n, 10)] =
                        (_Float16)x;
                }
            }
        }
    }
}

// ---------------------------------------------------------------------------
// Fused message-passing layer. One WG = 16 atom rows (tile i), 8 waves.
// h updated in place: h' = LN(h + softmax(MLP(rbf(d)))·t)
// tswz: f16 fragment-major 32-row chunks (from k_gemm swz=1).
// ---------------------------------------------------------------------------
__global__ __launch_bounds__(256)
void k_mp_attn(float* __restrict__ h, const _Float16* __restrict__ tswz,
               const float* __restrict__ pos, const unsigned char* __restrict__ mask,
               const float* __restrict__ aw1, const float* __restrict__ ab1,
               const float* __restrict__ aw2, const float* __restrict__ ab2,
               const float* __restrict__ nw, const float* __restrict__ nb,
               int Na) {
    __shared__ alignas(32) union { _Float16 lg[16 * 1536]; float msg[16 * HID]; } U;
    __shared__ alignas(32) union { _Float16 t_s[CH_HALFS]; float red[256]; } T;
    __shared__ alignas(32) _Float16 aw1s[32 * 48];
    __shared__ float ab1s[48], aw2s[48];
    __shared__ float px[16], py[16], pz[16], mki[16];
    __shared__ float rmax[16], rsum[16];

    int tiles = Na >> 4;
    int b  = blockIdx.x / tiles;
    int i0 = (blockIdx.x % tiles) << 4;
    int tid = threadIdx.x, lane = tid & 31, wv = tid >> 5;

    if (tid < 16) {
        int gi = b * Na + i0 + tid;
        px[tid] = pos[gi * 3 + 0];
        py[tid] = pos[gi * 3 + 1];
        pz[tid] = pos[gi * 3 + 2];
        mki[tid] = mask[gi] ? 1.0f : 0.0f;
    }
    for (int e = tid; e < 32 * 48; e += 256) {
        int k = e / 48, n = e % 48;
        aw1s[bswz(k, n, 3)] =
            (_Float16)((k < MBINS && n < AH) ? aw1[k * AH + n] : 0.0f);
    }
    if (tid < 48) {
        ab1s[tid] = (tid < AH) ? ab1[tid] : 0.0f;
        aw2s[tid] = (tid < AH) ? aw2[tid] : 0.0f;
    }
    __syncthreads();

    float ab2v = ab2[0];
    const float cstep = 10.0f / (float)(MBINS - 1);

    // weight fragments held in registers for the whole logit sweep
    AFrag awf[3];
    #pragma unroll
    for (int nt = 0; nt < 3; ++nt) awf[nt] = load_bfrag(aw1s + nt * 512);

    // ---- Phase 1: attention logits via WMMA (M = 16 i-rows per j column) ----
    int r = lane & 15, kb = (lane >> 4) * 8, ln = lane & 15;
    for (int jj = wv; jj < Na; jj += 8) {
        int gj = b * Na + jj;
        float jx = pos[gj * 3], jy = pos[gj * 3 + 1], jz = pos[gj * 3 + 2];
        float mj = mask[gj] ? 1.0f : 0.0f;
        float dx = px[r] - jx, dy = py[r] - jy, dz = pz[r] - jz;
        float d  = sqrtf(dx * dx + dy * dy + dz * dz);

        AFrag a;
        #pragma unroll
        for (int q = 0; q < 8; ++q) {
            int k0 = kb + q, k1 = kb + 16 + q;
            float u0 = d - cstep * (float)k0;
            float u1 = d - cstep * (float)k1;
            a.h[q]     = (k0 < MBINS) ? (_Float16)__expf(-2.0f * u0 * u0) : (_Float16)0.0f;
            a.h[8 + q] = (k1 < MBINS) ? (_Float16)__expf(-2.0f * u1 * u1) : (_Float16)0.0f;
        }

        float s[8];
        #pragma unroll
        for (int v = 0; v < 8; ++v) s[v] = 0.0f;
        #pragma unroll
        for (int nt = 0; nt < 3; ++nt) {
            CFrag c;
            #pragma unroll
            for (int v = 0; v < 8; ++v) c.f[v] = 0.0f;
            c.v = wmma16(a, awf[nt], c.v);
            float b1 = ab1s[nt * 16 + ln], w2 = aw2s[nt * 16 + ln];
            #pragma unroll
            for (int v = 0; v < 8; ++v) s[v] += gelu_f(c.f[v] + b1) * w2;
        }
        // reduce over N (half-wave butterfly, wave32)
        #pragma unroll
        for (int m = 1; m < 16; m <<= 1)
            #pragma unroll
            for (int v = 0; v < 8; ++v) s[v] += __shfl_xor(s[v], m, 32);

        float nbrf = (d < 10.0f && d > 0.01f) ? (mj * mki[r]) : 0.0f;
        #pragma unroll
        for (int v = 0; v < 8; ++v) {
            int row = v + 8 * (lane >> 4);
            float nv = __shfl(nbrf, row & 15, 32); // lanes (row&15) hold row's d
            float lv = (nv > 0.5f) ? (s[v] + ab2v) : -60000.0f;
            if ((lane & 15) == v) U.lg[row * 1536 + jj] = (_Float16)lv;
        }
    }
    __syncthreads();

    // ---- Phase 2: row max / sumexp ----
    int row2 = tid & 15, cix = tid >> 4;
    float mx = -1e30f;
    for (int j = cix; j < Na; j += 16) mx = fmaxf(mx, (float)U.lg[row2 * 1536 + j]);
    T.red[tid] = mx;
    __syncthreads();
    if (tid < 16) {
        float m = -1e30f;
        for (int q = 0; q < 16; ++q) m = fmaxf(m, T.red[tid + 16 * q]);
        rmax[tid] = m;
    }
    __syncthreads();
    float rm2 = rmax[row2], sm = 0.0f;
    for (int j = cix; j < Na; j += 16) sm += __expf((float)U.lg[row2 * 1536 + j] - rm2);
    __syncthreads();
    T.red[tid] = sm;
    __syncthreads();
    if (tid < 16) {
        float s2 = 0.0f;
        for (int q = 0; q < 16; ++q) s2 += T.red[tid + 16 * q];
        rsum[tid] = s2;
    }
    __syncthreads();

    // ---- Phase 3: msg = softmax_w @ t  (async-staged chunks + WMMA) ----
    CFrag acc[2];
    #pragma unroll
    for (int u = 0; u < 2; ++u)
        #pragma unroll
        for (int v = 0; v < 8; ++v) acc[u].f[v] = 0.0f;

    int nchunks = Na >> 5;
    float rmr = rmax[r];
    uint32_t tS_base = (uint32_t)(uintptr_t)(const void*)T.t_s;
    for (int jc = 0; jc < nchunks; ++jc) {
        __syncthreads();
        // verbatim 10KB chunk copy, no VGPR round trip (ASYNCcnt-tracked)
        const char* src = (const char*)(tswz + (size_t)(((b * Na) >> 5) + jc) * CH_HALFS);
        for (int e = tid; e < (CH_HALFS * 2) / 16; e += 256)
            async_copy_b128(tS_base + e * 16, src + e * 16);
        async_wait0();
        __syncthreads();

        AFrag a;
        #pragma unroll
        for (int q = 0; q < 8; ++q) {
            a.h[q]     = (_Float16)__expf((float)U.lg[r * 1536 + jc * 32 + kb + q]      - rmr);
            a.h[8 + q] = (_Float16)__expf((float)U.lg[r * 1536 + jc * 32 + kb + 16 + q] - rmr);
        }
        #pragma unroll
        for (int u = 0; u < 2; ++u) {
            int ht = wv + u * 8;
            if (ht < 10) {
                AFrag bf = load_bfrag(T.t_s + ht * 512);
                acc[u].v = wmma16(a, bf, acc[u].v);
            }
        }
    }
    __syncthreads(); // all lg reads done; union region reused as msg below

    // ---- Phase 4: normalize, residual, LayerNorm ----
    #pragma unroll
    for (int u = 0; u < 2; ++u) {
        int ht = wv + u * 8;
        if (ht < 10) {
            #pragma unroll
            for (int v = 0; v < 8; ++v) {
                int rr = v + 8 * (lane >> 4);
                U.msg[rr * HID + ht * 16 + ln] = acc[u].f[v] / rsum[rr];
            }
        }
    }
    __syncthreads();
    for (int e = tid; e < 16 * HID; e += 256) {
        int rr = e / HID, k = e % HID;
        U.msg[e] += h[(b * Na + i0 + rr) * HID + k];
    }
    __syncthreads();
    if (tid < 16) {
        float mean = 0.0f;
        for (int k = 0; k < HID; ++k) mean += U.msg[tid * HID + k];
        mean /= (float)HID;
        float var = 0.0f;
        for (int k = 0; k < HID; ++k) {
            float dd = U.msg[tid * HID + k] - mean;
            var += dd * dd;
        }
        var /= (float)HID;
        rmax[tid] = mean;
        rsum[tid] = rsqrtf(var + 1e-5f);
    }
    __syncthreads();
    for (int e = tid; e < 16 * HID; e += 256) {
        int rr = e / HID, k = e % HID;
        h[(b * Na + i0 + rr) * HID + k] =
            (U.msg[e] - rmax[rr]) * rsum[rr] * nw[k] + nb[k];
    }
}

// ---------------------------------------------------------------------------
// Fused cross-pair kernel. One WG per (b, l). Per 16-protein chunk:
// RBF WMMA (K=32 -> 80) builds pair tile; pe MLP WMMA (80->40->1).
// g_w rides in column 40 of pe_w1 so the gate logit falls out of the same
// WMMA C tile. Then row softmax over P and context accumulation.
// ---------------------------------------------------------------------------
__global__ __launch_bounds__(256)
void k_cross(const float* __restrict__ lig_pos, const float* __restrict__ prot_pos,
             const unsigned char* __restrict__ lig_mask,
             const unsigned char* __restrict__ prot_mask,
             const float* __restrict__ lpj, const float* __restrict__ pp,
             const float* __restrict__ rbf_w, const float* __restrict__ rbf_b,
             const float* __restrict__ pe_w1, const float* __restrict__ pe_b1,
             const float* __restrict__ pe_w2, const float* __restrict__ pe_b2,
             const float* __restrict__ g_w, const float* __restrict__ g_b,
             float* __restrict__ pe_out, float* __restrict__ ctx_out) {
    __shared__ alignas(32) _Float16 rbfw_s[32 * 80];
    __shared__ alignas(32) _Float16 pew1_s[96 * 48];
    __shared__ alignas(32) _Float16 pair_s[8][16 * 96];
    __shared__ float gl_s[1536];
    __shared__ float lpj_s[80], rbfb_s[80];
    __shared__ float peb1_s[48], pew2_s[48];
    __shared__ float red[256];

    int b = blockIdx.x / Ln, l = blockIdx.x % Ln;
    int tid = threadIdx.x, lane = tid & 31, wv = tid >> 5;

    for (int e = tid; e < 32 * 80; e += 256) {
        int k = e / 80, n = e % 80;
        rbfw_s[bswz(k, n, 5)] = (_Float16)rbf_w[e];
    }
    for (int e = tid; e < 96 * 48; e += 256) {
        int k = e / 48, n = e % 48;
        float w = 0.0f;
        if (k < PRD) {
            if (n < AH)       w = pe_w1[k * AH + n];
            else if (n == AH) w = g_w[k];   // gate weights in padded column 40
        }
        pew1_s[bswz(k, n, 3)] = (_Float16)w;
    }
    if (tid < 80) {
        lpj_s[tid]  = lpj[(b * Ln + l) * PRD + tid];
        rbfb_s[tid] = rbf_b[tid];
    }
    if (tid < 48) {
        peb1_s[tid] = (tid < AH) ? pe_b1[tid] : 0.0f;
        pew2_s[tid] = (tid < AH) ? pe_w2[tid] : 0.0f;
    }
    // zero pad cols 80..95 of this wave's pair tile
    for (int e = lane; e < 16 * 16; e += 32)
        pair_s[wv][(e >> 4) * 96 + 80 + (e & 15)] = (_Float16)0.0f;
    __syncthreads();

    // weight fragments in registers
    AFrag rbff[5], pef[9];
    #pragma unroll
    for (int nt = 0; nt < 5; ++nt) rbff[nt] = load_bfrag(rbfw_s + nt * 512);
    #pragma unroll
    for (int f = 0; f < 9; ++f) pef[f] = load_bfrag(pew1_s + f * 512);

    float lx = lig_pos[(b * Ln + l) * 3 + 0];
    float ly = lig_pos[(b * Ln + l) * 3 + 1];
    float lz = lig_pos[(b * Ln + l) * 3 + 2];
    float lmv = lig_mask[b * Ln + l] ? 1.0f : 0.0f;
    float peb2v = pe_b2[0], gbv = g_b[0];
    const float cstep = 10.0f / (float)(CBINS - 1);

    int r = lane & 15, kb = (lane >> 4) * 8, ln = lane & 15;
    for (int pc = wv; pc < (Pn >> 4); pc += 8) {
        if (pc + 8 < (Pn >> 4)) // prefetch next chunk's pp rows
            __builtin_prefetch(&pp[(b * Pn + (pc + 8) * 16 + r) * PRD], 0, 1);

        int pm = pc * 16 + r;
        int gp = b * Pn + pm;
        float dx = lx - prot_pos[gp * 3], dy = ly - prot_pos[gp * 3 + 1],
              dz = lz - prot_pos[gp * 3 + 2];
        float d = sqrtf(dx * dx + dy * dy + dz * dz);
        float cm_lane = (d < 10.0f && lmv > 0.5f && prot_mask[gp]) ? 1.0f : 0.0f;

        float cm_arr[8];
        #pragma unroll
        for (int v = 0; v < 8; ++v)
            cm_arr[v] = __shfl(cm_lane, (v + 8 * (lane >> 4)) & 15, 32);

        AFrag a;
        #pragma unroll
        for (int q = 0; q < 8; ++q) {
            float u0 = d - cstep * (float)(kb + q);
            float u1 = d - cstep * (float)(kb + 16 + q);
            a.h[q]     = (_Float16)__expf(-2.0f * u0 * u0);
            a.h[8 + q] = (_Float16)__expf(-2.0f * u1 * u1);
        }
        // rp = gelu(rbf @ rbf_w + rbf_b) * cm ; pair = lpj * pp * rp
        #pragma unroll
        for (int nt = 0; nt < 5; ++nt) {
            CFrag c;
            #pragma unroll
            for (int v = 0; v < 8; ++v) c.f[v] = 0.0f;
            c.v = wmma16(a, rbff[nt], c.v);
            int n = nt * 16 + ln;
            float rb = rbfb_s[n], lj = lpj_s[n];
            #pragma unroll
            for (int v = 0; v < 8; ++v) {
                int row = v + 8 * (lane >> 4);
                int pv = pc * 16 + row;
                float rp = gelu_f(c.f[v] + rb) * cm_arr[v];
                pair_s[wv][row * 96 + n] =
                    (_Float16)(rp * lj * pp[(b * Pn + pv) * PRD + n]);
            }
        }
        // pe = gelu(pair @ pe_w1 + pe_b1) @ pe_w2 + pe_b2 ; gl from column 40
        float s[8], glv[8];
        #pragma unroll
        for (int v = 0; v < 8; ++v) { s[v] = 0.0f; glv[v] = 0.0f; }
        #pragma unroll
        for (int nt = 0; nt < 3; ++nt) {
            CFrag c;
            #pragma unroll
            for (int v = 0; v < 8; ++v) c.f[v] = 0.0f;
            #pragma unroll
            for (int kc = 0; kc < 3; ++kc) {
                AFrag aa = load_a16(pair_s[wv] + kc * 32, 96);
                c.v = wmma16(aa, pef[kc * 3 + nt], c.v);
            }
            if (nt == 2) { // col 40 (ln==8) holds pair @ g_w
                #pragma unroll
                for (int v = 0; v < 8; ++v) glv[v] = c.f[v];
            }
            float b1 = peb1_s[nt * 16 + ln], w2 = pew2_s[nt * 16 + ln];
            #pragma unroll
            for (int v = 0; v < 8; ++v) s[v] += gelu_f(c.f[v] + b1) * w2;
        }
        #pragma unroll
        for (int m = 1; m < 16; m <<= 1)
            #pragma unroll
            for (int v = 0; v < 8; ++v) s[v] += __shfl_xor(s[v], m, 32);

        #pragma unroll
        for (int v = 0; v < 8; ++v) {
            float gv = __shfl(glv[v], 8 | (lane & 16), 32); // lanes 8 / 24
            if ((lane & 15) == v) {
                int row = v + 8 * (lane >> 4);
                int pv = pc * 16 + row;
                float cmv = cm_arr[v];
                pe_out[(b * Ln + l) * Pn + pv] = (s[v] + peb2v) * cmv;
                gl_s[pv] = (cmv > 0.5f) ? (gv + gbv) : -60000.0f;
            }
        }
    }
    __syncthreads();

    // softmax over P and context = sum_p w_p * pp[p,:]
    float mx = -1e30f;
    for (int p = tid; p < Pn; p += 256) mx = fmaxf(mx, gl_s[p]);
    red[tid] = mx;
    __syncthreads();
    for (int s2 = 128; s2 > 0; s2 >>= 1) {
        if (tid < s2) red[tid] = fmaxf(red[tid], red[tid + s2]);
        __syncthreads();
    }
    float gm = red[0];
    __syncthreads();
    float sm = 0.0f;
    for (int p = tid; p < Pn; p += 256) sm += __expf(gl_s[p] - gm);
    red[tid] = sm;
    __syncthreads();
    for (int s2 = 128; s2 > 0; s2 >>= 1) {
        if (tid < s2) red[tid] += red[tid + s2];
        __syncthreads();
    }
    float gs = red[0];
    __syncthreads();
    float acc = 0.0f;
    if (tid < 240) {
        int k = tid % 80, po = tid / 80;
        for (int p = po; p < Pn; p += 3)
            acc += (__expf(gl_s[p] - gm) / gs) * pp[(b * Pn + p) * PRD + k];
    }
    red[tid] = (tid < 240) ? acc : 0.0f;
    __syncthreads();
    if (tid < 80)
        ctx_out[(b * Ln + l) * PRD + tid] = red[tid] + red[tid + 80] + red[tid + 160];
}

// ---------------------------------------------------------------------------
// lhc = LN(lh + ctx@c_w + c_b) * lig_mask ; accumulate sums for complex_repr
// ---------------------------------------------------------------------------
__global__ __launch_bounds__(160)
void k_lhc(const float* __restrict__ lh, const float* __restrict__ cproj,
           const unsigned char* __restrict__ lig_mask,
           const float* __restrict__ cn_w, const float* __restrict__ cn_b,
           float* __restrict__ comp_sum) {
    __shared__ float x[HID];
    __shared__ float st[2];
    int b = blockIdx.x / Ln, l = blockIdx.x % Ln;
    int row = b * Ln + l, tid = threadIdx.x;
    if (tid < HID) x[tid] = lh[row * HID + tid] + cproj[row * HID + tid];
    __syncthreads();
    if (tid == 0) {
        float m = 0.0f;
        for (int k = 0; k < HID; ++k) m += x[k];
        m /= (float)HID;
        float v = 0.0f;
        for (int k = 0; k < HID; ++k) { float d = x[k] - m; v += d * d; }
        st[0] = m;
        st[1] = rsqrtf(v / (float)HID + 1e-5f);
    }
    __syncthreads();
    float mk = lig_mask[row] ? 1.0f : 0.0f;
    if (tid < HID) {
        float val = ((x[tid] - st[0]) * st[1] * cn_w[tid] + cn_b[tid]) * mk;
        atomicAdd(&comp_sum[b * HID + tid], val);
    }
}

__global__ void k_zero(float* p, int n) {
    int i = blockIdx.x * blockDim.x + threadIdx.x;
    if (i < n) p[i] = 0.0f;
}

__global__ __launch_bounds__(256)
void k_reduce_pe(const float* __restrict__ pe, float* __restrict__ tpe) {
    __shared__ float red[256];
    int b = blockIdx.x, tid = threadIdx.x;
    float s = 0.0f;
    const float* p = pe + b * Ln * Pn;
    for (int i = tid; i < Ln * Pn; i += 256) s += p[i];
    red[tid] = s;
    __syncthreads();
    for (int s2 = 128; s2 > 0; s2 >>= 1) {
        if (tid < s2) red[tid] += red[tid + s2];
        __syncthreads();
    }
    if (tid == 0) tpe[b] = red[0];
}

__global__ __launch_bounds__(256)
void k_head(const float* __restrict__ csum, const float* __restrict__ tpe,
            const unsigned char* __restrict__ lig_mask,
            const float* __restrict__ a_w1, const float* __restrict__ a_b1,
            const float* __restrict__ a_w2, const float* __restrict__ a_b2,
            const float* __restrict__ cf_w1, const float* __restrict__ cf_b1,
            const float* __restrict__ cf_w2, const float* __restrict__ cf_b2,
            const float* __restrict__ scale, const float* __restrict__ bias,
            float* __restrict__ out) {
    __shared__ float repr[HID];
    __shared__ float red[256];
    __shared__ float nlig;
    int b = blockIdx.x, tid = threadIdx.x;
    if (tid == 0) {
        int s = 0;
        for (int i = 0; i < Ln; ++i) s += lig_mask[b * Ln + i] ? 1 : 0;
        nlig = fmaxf((float)s, 1.0f);
    }
    __syncthreads();
    if (tid < HID) repr[tid] = csum[b * HID + tid] / nlig;
    __syncthreads();
    float pa = 0.0f, pc = 0.0f;
    if (tid < PRD) {
        float sa = a_b1[tid], sc = cf_b1[tid];
        for (int k = 0; k < HID; ++k) {
            float rr = repr[k];
            sa += rr * a_w1[k * PRD + tid];
            sc += rr * cf_w1[k * PRD + tid];
        }
        pa = gelu_f(sa) * a_w2[tid];
        pc = gelu_f(sc) * cf_w2[tid];
    }
    red[tid] = pa;
    __syncthreads();
    for (int s2 = 128; s2 > 0; s2 >>= 1) {
        if (tid < s2) red[tid] += red[tid + s2];
        __syncthreads();
    }
    float ha = red[0];
    __syncthreads();
    red[tid] = pc;
    __syncthreads();
    for (int s2 = 128; s2 > 0; s2 >>= 1) {
        if (tid < s2) red[tid] += red[tid + s2];
        __syncthreads();
    }
    float hc = red[0];
    if (tid == 0) {
        out[b] = tpe[b] * scale[0] + ha + a_b2[0] + bias[0];
        float z = hc + cf_b2[0];
        out[Bn + b] = 1.0f / (1.0f + __expf(-z));
    }
}

// ---------------------------------------------------------------------------
// Host-side launch
// ---------------------------------------------------------------------------
enum {
    IN_PROT_POS = 0, IN_PROT_FEAT, IN_LIG_POS, IN_LIG_FEAT,
    IN_PROT_MASK, IN_LIG_MASK,
    // params, sorted pytree key order
    P_A_B1 = 6, P_A_B2, P_A_W1, P_A_W2, P_BIAS, P_C_B, P_C_W,
    P_CF_B1, P_CF_B2, P_CF_W1, P_CF_W2, P_CN_B, P_CN_W, P_G_B, P_G_W,
    P_L_EB1, P_L_EB2, P_L_EW1, P_L_EW2,
    P_LIG_L0 = 25,   // 2 layers x 8: ab1,ab2,aw1,aw2,mb,mw,nb,nw
    P_LP_B = 41, P_LP_W, P_P_EB1, P_P_EB2, P_P_EW1, P_P_EW2,
    P_PE_B1 = 47, P_PE_B2, P_PE_W1, P_PE_W2, P_PP_B, P_PP_W,
    P_PROT_L0 = 53,  // 4 layers x 8
    P_RBF_B = 85, P_RBF_W, P_SCALE
};

extern "C" void kernel_launch(void* const* d_in, const int* in_sizes, int n_in,
                              void* d_out, int out_size, void* d_ws, size_t ws_size,
                              hipStream_t stream) {
    auto F = [&](int i) { return (const float*)d_in[i]; };
    const float* prot_pos  = F(IN_PROT_POS);
    const float* prot_feat = F(IN_PROT_FEAT);
    const float* lig_pos   = F(IN_LIG_POS);
    const float* lig_feat  = F(IN_LIG_FEAT);
    const unsigned char* prot_mask = (const unsigned char*)d_in[IN_PROT_MASK];
    const unsigned char* lig_mask  = (const unsigned char*)d_in[IN_LIG_MASK];

    float* ws   = (float*)d_ws;
    float* ph   = ws;                    // [B*P,160] f32
    float* tb   = ph   + Bn * Pn * HID;  // t: f16 swizzled chunks (or f32 scratch)
    float* lhp  = tb   + Bn * Pn * HID;  // [B*L,160] f32
    float* ltb  = lhp  + Bn * Ln * HID;  // ligand t: f16 swizzled chunks / scratch
    float* ppb  = ltb  + Bn * Ln * HID;  // [B*P,80]
    float* lpjb = ppb  + Bn * Pn * PRD;  // [B*L,80]
    float* ctx  = lpjb + Bn * Ln * PRD;  // [B*L,80]
    float* cpj  = ctx  + Bn * Ln * PRD;  // [B*L,160]
    float* csum = cpj  + Bn * Ln * HID;  // [B,160]
    float* tpe  = csum + Bn * HID;       // [B]

    float* out    = (float*)d_out;
    float* pe_out = out + 2 * Bn; // pkd[B], conf[B], pe[B,L,P]

    const int MP = Bn * Pn, ML = Bn * Ln;

    // protein embed (tb used as plain f32 scratch here)
    k_gemm<<<MP / 16, 256, 0, stream>>>(prot_feat, F(P_P_EW1), F(P_P_EB1), tb, MP, NFT, HID, 1, 0);
    k_gemm<<<MP / 16, 256, 0, stream>>>(tb, F(P_P_EW2), F(P_P_EB2), ph, MP, HID, HID, 0, 0);
    for (int lyr = 0; lyr < 4; ++lyr) {
        int base = P_PROT_L0 + 8 * lyr;
        // t = gelu(h@mw+mb), written as f16 fragment-major chunks
        k_gemm<<<MP / 16, 256, 0, stream>>>(ph, F(base + 5), F(base + 4), tb, MP, HID, HID, 1, 1);
        k_mp_attn<<<MP / 16, 256, 0, stream>>>(ph, (const _Float16*)tb, prot_pos, prot_mask,
            F(base + 2), F(base + 0), F(base + 3), F(base + 1), F(base + 7), F(base + 6), Pn);
    }
    // ligand embed
    k_gemm<<<ML / 16, 256, 0, stream>>>(lig_feat, F(P_L_EW1), F(P_L_EB1), ltb, ML, NFT, HID, 1, 0);
    k_gemm<<<ML / 16, 256, 0, stream>>>(ltb, F(P_L_EW2), F(P_L_EB2), lhp, ML, HID, HID, 0, 0);
    for (int lyr = 0; lyr < 2; ++lyr) {
        int base = P_LIG_L0 + 8 * lyr;
        k_gemm<<<ML / 16, 256, 0, stream>>>(lhp, F(base + 5), F(base + 4), ltb, ML, HID, HID, 1, 1);
        k_mp_attn<<<ML / 16, 256, 0, stream>>>(lhp, (const _Float16*)ltb, lig_pos, lig_mask,
            F(base + 2), F(base + 0), F(base + 3), F(base + 1), F(base + 7), F(base + 6), Ln);
    }
    // pair projections
    k_gemm<<<MP / 16, 256, 0, stream>>>(ph, F(P_PP_W), F(P_PP_B), ppb, MP, HID, PRD, 0, 0);
    k_gemm<<<ML / 16, 256, 0, stream>>>(lhp, F(P_LP_W), F(P_LP_B), lpjb, ML, HID, PRD, 0, 0);
    // cross pair energies, gate softmax, context
    k_cross<<<Bn * Ln, 256, 0, stream>>>(lig_pos, prot_pos, lig_mask, prot_mask,
        lpjb, ppb, F(P_RBF_W), F(P_RBF_B), F(P_PE_W1), F(P_PE_B1), F(P_PE_W2), F(P_PE_B2),
        F(P_G_W), F(P_G_B), pe_out, ctx);
    // context projection + lhc LN + complex_repr sums
    k_gemm<<<ML / 16, 256, 0, stream>>>(ctx, F(P_C_W), F(P_C_B), cpj, ML, PRD, HID, 0, 0);
    k_zero<<<2, 256, 0, stream>>>(csum, Bn * HID + Bn);
    k_lhc<<<Bn * Ln, 160, 0, stream>>>(lhp, cpj, lig_mask, F(P_CN_W), F(P_CN_B), csum);
    k_reduce_pe<<<Bn, 256, 0, stream>>>(pe_out, tpe);
    k_head<<<Bn, 256, 0, stream>>>(csum, tpe, lig_mask,
        F(P_A_W1), F(P_A_B1), F(P_A_W2), F(P_A_B2),
        F(P_CF_W1), F(P_CF_B1), F(P_CF_W2), F(P_CF_B2),
        F(P_SCALE), F(P_BIAS), out);
}
